// MambaBlock_43980465111222
// MI455X (gfx1250) — matrix operand order repaired and checked by reading, not compiled, not run on hardware
//
#include <hip/hip_runtime.h>

typedef __attribute__((ext_vector_type(16))) _Float16 v16h;
typedef __attribute__((ext_vector_type(8)))  _Float16 v8h;
typedef __attribute__((ext_vector_type(8)))  float    v8f;

union V16 { v16h v; v8h h[2]; };

#define D_MODEL 1024
#define D_STATE 16
#define D_CONV  4
#define D_INNER 2048
#define BATCH   2
#define SEQ     2048
#define MROWS   (BATCH * SEQ)   // 4096
#define NXZ     (2 * D_INNER)   // 4096

// ---------------------------------------------------------------------------
// CDNA5 async global->LDS copy (16B per lane), tracked by ASYNCcnt.
// LDS address = low 32 bits of the generic pointer (HW truncates flat LDS
// addresses to the wave-relative LDS offset).
// ---------------------------------------------------------------------------
__device__ __forceinline__ void async_ld_b128(const _Float16* lds_dst,
                                              const _Float16* gsrc) {
  unsigned lds = (unsigned)(unsigned long long)(uintptr_t)lds_dst;
  asm volatile("global_load_async_to_lds_b128 %0, %1, off"
               :: "v"(lds), "v"((unsigned long long)(uintptr_t)gsrc)
               : "memory");
}
__device__ __forceinline__ void wait_async0() {
  asm volatile("s_wait_asynccnt 0x0" ::: "memory");
}

// ---------------------------------------------------------------------------
// f32 -> f16 conversion
// ---------------------------------------------------------------------------
__global__ void cvt_f16_kernel(const float* __restrict__ in,
                               _Float16* __restrict__ out, int n) {
  int i = blockIdx.x * blockDim.x + threadIdx.x;
  if (i < n) out[i] = (_Float16)in[i];
}

// f32 (KxN row-major) -> f16 transposed (NxK row-major)
__global__ void cvt_f16_tr_kernel(const float* __restrict__ in,
                                  _Float16* __restrict__ out, int K, int N) {
  int i = blockIdx.x * blockDim.x + threadIdx.x;
  if (i < K * N) {
    int k = i / N, n = i % N;
    out[(size_t)n * K + k] = (_Float16)in[i];
  }
}

// ---------------------------------------------------------------------------
// WMMA GEMM: C(MxN,f32) = A(MxK,f16,row-major) @ Bt(NxK,f16,row-major)^T
// 256 threads = 8 waves. Block tile 128x128, wave tile 64x32 (4x2 WMMA accs).
// Double-buffered LDS staging via global_load_async_to_lds_b128: tile k+1's
// DMA runs concurrently with tile k's WMMAs; one barrier per K-step.
// ---------------------------------------------------------------------------
__global__ __launch_bounds__(256) void wmma_gemm_f16f32_kernel(
    const _Float16* __restrict__ A, const _Float16* __restrict__ Bt,
    float* __restrict__ C, int Mdim, int Ndim, int Kdim) {
  (void)Mdim;
  __shared__ _Float16 As[2][128 * 32];  // As[p][row*32 + k]
  __shared__ _Float16 Bs[2][128 * 32];  // Bs[p][col*32 + k]

  const int tid   = threadIdx.x;
  const int lane  = tid & 31;
  const int wid   = tid >> 5;
  const int waveM = wid >> 2;  // 0..1
  const int waveN = wid & 3;   // 0..3
  const int l16   = lane & 15;
  const int hi    = lane >> 4; // 0 or 1
  const int klo   = hi * 8;    // A-fragment K base (ISA 16-bit A layout)
  const int kb    = hi * 16;   // unused; B fragment also uses klo
  const int blockM = blockIdx.y * 128;
  const int blockN = blockIdx.x * 128;

  // chunk decomposition: 512 chunks of 8 halves per tile, 2 per thread
  const int c0row = tid >> 2;               // 0..63
  const int c0col = (tid & 3) * 8;          // 0,8,16,24
  const int c1row = c0row + 64;             // 64..127

  auto stage = [&](int kk, int pp) {
    async_ld_b128(&As[pp][c0row * 32 + c0col],
                  A + (size_t)(blockM + c0row) * Kdim + kk + c0col);
    async_ld_b128(&As[pp][c1row * 32 + c0col],
                  A + (size_t)(blockM + c1row) * Kdim + kk + c0col);
    async_ld_b128(&Bs[pp][c0row * 32 + c0col],
                  Bt + (size_t)(blockN + c0row) * Kdim + kk + c0col);
    async_ld_b128(&Bs[pp][c1row * 32 + c0col],
                  Bt + (size_t)(blockN + c1row) * Kdim + kk + c0col);
  };

  v8f acc[4][2];
#pragma unroll
  for (int mt = 0; mt < 4; ++mt)
#pragma unroll
    for (int nt = 0; nt < 2; ++nt) acc[mt][nt] = (v8f){};

  // prologue: stage tile 0 into buffer 0
  stage(0, 0);
  wait_async0();
  __syncthreads();

  int p = 0;
  for (int k0 = 0; k0 < Kdim; k0 += 32) {
    // overlap: DMA tile k+1 into the other buffer while computing tile k
    if (k0 + 32 < Kdim) stage(k0 + 32, 1 - p);
    // prime L2 two tiles ahead (gfx1250 global_prefetch_b8)
    if (k0 + 64 < Kdim) {
      __builtin_prefetch(A + (size_t)(blockM + c0row) * Kdim + (k0 + 64) + c0col, 0, 0);
      __builtin_prefetch(Bt + (size_t)(blockN + c0row) * Kdim + (k0 + 64) + c0col, 0, 0);
    }

    // --- load fragments of tile k from LDS ---
    v16h af[4], bf[2];
#pragma unroll
    for (int mt = 0; mt < 4; ++mt) {
      const _Float16* q = &As[p][(waveM * 64 + mt * 16 + l16) * 32];
      V16 u;
      u.h[0] = *(const v8h*)(q + klo);       // K = klo .. klo+7
      u.h[1] = *(const v8h*)(q + 16 + klo);  // K = 16+klo .. 16+klo+7
      af[mt] = u.v;
    }
#pragma unroll
    for (int nt = 0; nt < 2; ++nt) {
      const _Float16* q = &Bs[p][(waveN * 32 + nt * 16 + l16) * 32];
      V16 u;
      u.h[0] = *(const v8h*)(q + klo);       // K = klo .. klo+7
      u.h[1] = *(const v8h*)(q + 16 + klo);  // K = 16+klo .. 16+klo+7
      bf[nt] = u.v;
    }

    // --- 8 WMMAs per K-step (overlapped with next tile's DMA) ---
#pragma unroll
    for (int mt = 0; mt < 4; ++mt)
#pragma unroll
      for (int nt = 0; nt < 2; ++nt)
        acc[mt][nt] = __builtin_amdgcn_wmma_f32_16x16x32_f16(
            false, af[mt], false, bf[nt], (short)0, acc[mt][nt], false, false);

    // next tile's DMA must land, and all waves' reads of buf p are done
    wait_async0();
    __syncthreads();
    p ^= 1;
  }

  // --- epilogue: C/D layout -> VGPR r: lanes0-15 M=r, lanes16-31 M=8+r ---
#pragma unroll
  for (int mt = 0; mt < 4; ++mt)
#pragma unroll
    for (int nt = 0; nt < 2; ++nt) {
      int col = blockN + waveN * 32 + nt * 16 + l16;
#pragma unroll
      for (int r = 0; r < 8; ++r) {
        int row = blockM + waveM * 64 + mt * 16 + hi * 8 + r;
        C[(size_t)row * Ndim + col] = acc[mt][nt][r];
      }
    }
}

// ---------------------------------------------------------------------------
// Depthwise causal conv (K=4) + bias + SiLU over the xa half of xz
// grid: (D_INNER/256, SEQ, BATCH)
// ---------------------------------------------------------------------------
__global__ void conv_silu_kernel(const float* __restrict__ xz,
                                 const float* __restrict__ conv_w,
                                 const float* __restrict__ conv_b,
                                 float* __restrict__ xc) {
  const int d = blockIdx.x * 256 + threadIdx.x;
  const int l = blockIdx.y;
  const int b = blockIdx.z;
  const int m = b * SEQ + l;
  float a = conv_b[d];
#pragma unroll
  for (int j = 0; j < D_CONV; ++j) {
    int ls = l - (D_CONV - 1) + j;
    if (ls >= 0)
      a += conv_w[d * D_CONV + j] * xz[(size_t)(b * SEQ + ls) * NXZ + d];
  }
  xc[(size_t)m * D_INNER + d] = a / (1.f + __expf(-a));  // silu
}

// ---------------------------------------------------------------------------
// BC = xc @ W_xproj (K=2048, N=32): one wave per row, lane = output column
// ---------------------------------------------------------------------------
__global__ void xproj_kernel(const float* __restrict__ xc,
                             const float* __restrict__ W_xproj,
                             float* __restrict__ BC) {
  const int lane = threadIdx.x & 31;
  const int m = blockIdx.x * 8 + (threadIdx.x >> 5);
  const float* xr = xc + (size_t)m * D_INNER;
  float a = 0.f;
  for (int dd = 0; dd < D_INNER; ++dd) a += xr[dd] * W_xproj[dd * 32 + lane];
  BC[m * 32 + lane] = a;
}

// ---------------------------------------------------------------------------
// delta = softplus(B_ssm @ W_dt + b_dt)   (K=16)
// grid: (D_INNER/256, SEQ, BATCH)
// ---------------------------------------------------------------------------
__global__ void delta_kernel(const float* __restrict__ BC,
                             const float* __restrict__ W_dt,
                             const float* __restrict__ b_dt,
                             float* __restrict__ dl) {
  const int d = blockIdx.x * 256 + threadIdx.x;
  const int m = blockIdx.z * SEQ + blockIdx.y;
  float a = b_dt[d];
#pragma unroll
  for (int s = 0; s < D_STATE; ++s) a += BC[m * 32 + s] * W_dt[s * D_INNER + d];
  dl[(size_t)m * D_INNER + d] = (a > 20.f) ? a : log1pf(__expf(a));
}

// ---------------------------------------------------------------------------
// Selective scan, fused with +D*x, silu(z) gating and f16 cast of the result.
// One thread per (batch, channel); h[16] in registers; per-step B/C via LDS.
// ---------------------------------------------------------------------------
__global__ __launch_bounds__(256) void scan_kernel(
    const float* __restrict__ xc, const float* __restrict__ dl,
    const float* __restrict__ BC, const float* __restrict__ xz,
    const float* __restrict__ A_log, const float* __restrict__ Dp,
    _Float16* __restrict__ G) {
  const int d = blockIdx.x * 256 + threadIdx.x;
  const int b = blockIdx.y;
  float h[D_STATE], a[D_STATE];
#pragma unroll
  for (int s = 0; s < D_STATE; ++s) {
    h[s] = 0.f;
    a[s] = -__expf(A_log[d * D_STATE + s]);
  }
  const float Dd = Dp[d];
  __shared__ float sBC[32];
  for (int l = 0; l < SEQ; ++l) {
    const int m = b * SEQ + l;
    __syncthreads();
    if (threadIdx.x < 32) sBC[threadIdx.x] = BC[m * 32 + threadIdx.x];
    __syncthreads();
    const float dt = dl[(size_t)m * D_INNER + d];
    const float xt = xc[(size_t)m * D_INNER + d];
    const float dx = dt * xt;
    float y = 0.f;
#pragma unroll
    for (int s = 0; s < D_STATE; ++s) {
      const float dA = __expf(dt * a[s]);
      h[s] = dA * h[s] + dx * sBC[s];
      y += h[s] * sBC[16 + s];
    }
    const float z = xz[(size_t)m * NXZ + D_INNER + d];
    const float sz = z / (1.f + __expf(-z));
    G[(size_t)m * D_INNER + d] = (_Float16)((y + Dd * xt) * sz);
  }
}

// ---------------------------------------------------------------------------
extern "C" void kernel_launch(void* const* d_in, const int* in_sizes, int n_in,
                              void* d_out, int out_size, void* d_ws, size_t ws_size,
                              hipStream_t stream) {
  (void)in_sizes; (void)n_in; (void)out_size; (void)ws_size;
  const float* x       = (const float*)d_in[0];
  const float* W_in    = (const float*)d_in[1];
  const float* conv_w  = (const float*)d_in[2];
  const float* conv_b  = (const float*)d_in[3];
  const float* W_xproj = (const float*)d_in[4];
  const float* W_dt    = (const float*)d_in[5];
  const float* b_dt    = (const float*)d_in[6];
  const float* A_log   = (const float*)d_in[7];
  const float* Dp      = (const float*)d_in[8];
  const float* W_out   = (const float*)d_in[9];
  float* out = (float*)d_out;
  char* ws = (char*)d_ws;

  size_t off = 0;
  float*    xz   = (float*)(ws + off);    off += (size_t)MROWS * NXZ * 4;       // 64 MB
  float*    xc   = (float*)(ws + off);    off += (size_t)MROWS * D_INNER * 4;   // 32 MB
  float*    dl   = (float*)(ws + off);    off += (size_t)MROWS * D_INNER * 4;   // 32 MB
  float*    BC   = (float*)(ws + off);    off += (size_t)MROWS * 32 * 4;        // 0.5 MB
  _Float16* xh   = (_Float16*)(ws + off); off += (size_t)MROWS * D_MODEL * 2;   // 8 MB
  _Float16* WihT = (_Float16*)(ws + off); off += (size_t)D_MODEL * NXZ * 2;     // 8 MB  (NXZ x D_MODEL)
  _Float16* WohT = (_Float16*)(ws + off); off += (size_t)D_INNER * D_MODEL * 2; // 4 MB  (D_MODEL x D_INNER)
  _Float16* G    = (_Float16*)(ws + off); off += (size_t)MROWS * D_INNER * 2;   // 16 MB

  // 1) convert GEMM operands to f16 (weights transposed to NxK for async staging)
  {
    int n = MROWS * D_MODEL;
    cvt_f16_kernel<<<(n + 255) / 256, 256, 0, stream>>>(x, xh, n);
    n = D_MODEL * NXZ;
    cvt_f16_tr_kernel<<<(n + 255) / 256, 256, 0, stream>>>(W_in, WihT, D_MODEL, NXZ);
    n = D_INNER * D_MODEL;
    cvt_f16_tr_kernel<<<(n + 255) / 256, 256, 0, stream>>>(W_out, WohT, D_INNER, D_MODEL);
  }
  // 2) xz = x @ W_in  (4096x1024 @ 1024x4096)
  wmma_gemm_f16f32_kernel<<<dim3(NXZ / 128, MROWS / 128), 256, 0, stream>>>(
      xh, WihT, xz, MROWS, NXZ, D_MODEL);
  // 3) depthwise causal conv + SiLU
  conv_silu_kernel<<<dim3(D_INNER / 256, SEQ, BATCH), 256, 0, stream>>>(xz, conv_w, conv_b, xc);
  // 4) BC = xc @ W_xproj
  xproj_kernel<<<MROWS / 8, 256, 0, stream>>>(xc, W_xproj, BC);
  // 5) delta = softplus(B @ W_dt + b_dt)
  delta_kernel<<<dim3(D_INNER / 256, SEQ, BATCH), 256, 0, stream>>>(BC, W_dt, b_dt, dl);
  // 6) selective scan + D*x + silu(z) gate, output f16 G
  scan_kernel<<<dim3(D_INNER / 256, BATCH), 256, 0, stream>>>(xc, dl, BC, xz, A_log, Dp, G);
  // 7) out = G @ W_out  (4096x2048 @ 2048x1024)
  wmma_gemm_f16f32_kernel<<<dim3(D_MODEL / 128, MROWS / 128), 256, 0, stream>>>(
      G, WohT, out, MROWS, D_MODEL, D_INNER);
}
